// QKVAttentionLegacy_43327630082332
// MI455X (gfx1250) — matrix operand order repaired and checked by reading, not compiled, not run on hardware
//
#include <hip/hip_runtime.h>

typedef __attribute__((ext_vector_type(16))) _Float16 v16h;
typedef __attribute__((ext_vector_type(8)))  _Float16 v8h;
typedef __attribute__((ext_vector_type(8)))  float    v8f;

#define SEQ   2048
#define CH    64
#define BT    128      // query rows per block (8 waves * 16)
#define BS    64       // key/value rows per iteration
#define NW    8        // waves per block
#define PITCH 72       // halves per LDS row: 16B-aligned rows + bank spread

union V16U { v16h v; v8h h[2]; };

__global__ __launch_bounds__(256)
void qkv_attn_flash_f16wmma(const float* __restrict__ qkv, float* __restrict__ out)
{
    const int bh   = blockIdx.y;            // 0..63  (bs*heads)
    const int tblk = blockIdx.x;            // 0..15
    const int tid  = threadIdx.x;
    const int wv   = tid >> 5;
    const int ln   = tid & 31;
    const int lh   = ln >> 4;               // which 16-lane half
    const int l15  = ln & 15;

    const float* qp = qkv + (size_t)bh * 192 * SEQ;          // q rows 0..63
    const float* kp = qp + (size_t)CH * SEQ;                 // k rows 64..127
    const float* vp = qp + (size_t)(2 * CH) * SEQ;           // v rows 128..191

    __shared__ __align__(16) _Float16 sK[BS * PITCH];        // K^T : [s][c]
    __shared__ __align__(16) _Float16 sV[CH * PITCH];        // V   : [c][s]
    __shared__ __align__(16) _Float16 sP[NW * 16 * PITCH];   // per-wave P tile [t][s]
    __shared__ float sStat[NW * 16];                         // per-wave row factors

    const int tWave = tblk * BT + wv * 16;                   // wave's first t
    const float scale = 0.35355339059327373f;                // 64^(-1/4)

    // ---- Q tile -> WMMA A-layout registers (K = c, two 32-wide chunks) ----
    V16U qa[2];
    #pragma unroll
    for (int ck = 0; ck < 2; ++ck) {
        #pragma unroll
        for (int e = 0; e < 16; ++e) {
            int c = ck * 32 + lh * 8 + (e < 8 ? e : e + 8);
            qa[ck].v[e] = (_Float16)(qp[(size_t)c * SEQ + tWave + l15] * scale);
        }
    }

    // O accumulator: D of GEMM2 (m = c over 4 tiles, n = t)
    v8f acc[4];
    #pragma unroll
    for (int i = 0; i < 4; ++i)
        #pragma unroll
        for (int r = 0; r < 8; ++r) acc[i][r] = 0.0f;

    float mrun[8], lrun[8];                  // stats for row t = r + lh*8
    #pragma unroll
    for (int r = 0; r < 8; ++r) { mrun[r] = -1e30f; lrun[r] = 0.0f; }

    _Float16* myP   = sP + wv * 16 * PITCH;
    float*    mySt  = sStat + wv * 16;

    #pragma unroll 1
    for (int s0 = 0; s0 < SEQ; s0 += BS) {
        __syncthreads();                     // previous-iter LDS reads done
        // ---- cooperative K/V block load, f32 -> f16, K transposed ----
        {
            int sL = (tid & 15) * 4;         // 0..60, float4 granularity
            int cB = tid >> 4;               // 0..15
            #pragma unroll
            for (int cc = 0; cc < 4; ++cc) {
                int c = cB + cc * 16;
                float4 k4 = *(const float4*)(kp + (size_t)c * SEQ + s0 + sL);
                float4 v4 = *(const float4*)(vp + (size_t)c * SEQ + s0 + sL);
                sK[(sL + 0) * PITCH + c] = (_Float16)(k4.x * scale);
                sK[(sL + 1) * PITCH + c] = (_Float16)(k4.y * scale);
                sK[(sL + 2) * PITCH + c] = (_Float16)(k4.z * scale);
                sK[(sL + 3) * PITCH + c] = (_Float16)(k4.w * scale);
                _Float16* vd = sV + c * PITCH + sL;
                vd[0] = (_Float16)v4.x; vd[1] = (_Float16)v4.y;
                vd[2] = (_Float16)v4.z; vd[3] = (_Float16)v4.w;
            }
        }
        __syncthreads();

        // ---- GEMM1: S = Q^T K  (4 s-tiles of 16x16, K-dim = 64) ----
        v8f S[4];
        #pragma unroll
        for (int j = 0; j < 4; ++j) {
            v8f cz = {};
            #pragma unroll
            for (int kk = 0; kk < 2; ++kk) {
                // B layout: n = s = j*16+l15 (row of sK), K = c = kk*32 + lh*8 + (e<8?e:e+8)
                const _Float16* bp = sK + (j * 16 + l15) * PITCH + kk * 32 + lh * 8;
                V16U b;
                b.h[0] = *(const v8h*)(bp);
                b.h[1] = *(const v8h*)(bp + 16);
                cz = __builtin_amdgcn_wmma_f32_16x16x32_f16(
                        false, qa[kk].v, false, b.v, (short)0, cz, false, false);
            }
            S[j] = cz;
        }

        // ---- online softmax (row t = r + lh*8; N spread over lanes 0..15) ----
        float corr[8];
        #pragma unroll
        for (int r = 0; r < 8; ++r) {
            float bm = fmaxf(fmaxf(S[0][r], S[1][r]), fmaxf(S[2][r], S[3][r]));
            #pragma unroll
            for (int off = 8; off >= 1; off >>= 1)
                bm = fmaxf(bm, __shfl_xor(bm, off, 32));
            float mn = fmaxf(mrun[r], bm);
            corr[r] = __expf(mrun[r] - mn);
            mrun[r] = mn;
            float rs = 0.0f;
            #pragma unroll
            for (int j = 0; j < 4; ++j) {
                float p = __expf(S[j][r] - mn);
                S[j][r] = p;
                rs += p;
            }
            #pragma unroll
            for (int off = 8; off >= 1; off >>= 1)
                rs += __shfl_xor(rs, off, 32);
            lrun[r] = lrun[r] * corr[r] + rs;
        }

        // ---- P (D-layout) -> per-wave LDS [t][s] as f16; export row factors ----
        #pragma unroll
        for (int j = 0; j < 4; ++j)
            #pragma unroll
            for (int r = 0; r < 8; ++r)
                myP[(r + lh * 8) * PITCH + j * 16 + l15] = (_Float16)S[j][r];
        if (l15 == 0) {
            #pragma unroll
            for (int r = 0; r < 8; ++r) mySt[r + lh * 8] = corr[r];
        }
        __builtin_amdgcn_wave_barrier();     // order per-wave LDS WAR/RAW

        // ---- rescale O, GEMM2: O^T += V * P^T ----
        float cf = mySt[l15];                // factor for t = l15
        #pragma unroll
        for (int ct = 0; ct < 4; ++ct)
            #pragma unroll
            for (int r = 0; r < 8; ++r) acc[ct][r] *= cf;

        #pragma unroll
        for (int kk = 0; kk < 2; ++kk) {
            // B = P^T: n = t = l15 (row of myP), K = s = kk*32 + lh*8 + (e<8?e:e+8)
            const _Float16* bp = myP + l15 * PITCH + kk * 32 + lh * 8;
            V16U b;
            b.h[0] = *(const v8h*)(bp);
            b.h[1] = *(const v8h*)(bp + 16);
            #pragma unroll
            for (int ct = 0; ct < 4; ++ct) {
                // A = V: m = c = ct*16 + l15, K = s = kk*32 + lh*8 + (e<8?e:e+8)
                const _Float16* ap = sV + (ct * 16 + l15) * PITCH + kk * 32 + lh * 8;
                V16U a;
                a.h[0] = *(const v8h*)(ap);
                a.h[1] = *(const v8h*)(ap + 16);
                acc[ct] = __builtin_amdgcn_wmma_f32_16x16x32_f16(
                            false, a.v, false, b.v, (short)0, acc[ct], false, false);
            }
        }
    }

    // ---- finalize: O /= l, store out[c][t] (coalesced in t) ----
    __syncthreads();
    if (l15 == 0) {
        #pragma unroll
        for (int r = 0; r < 8; ++r) mySt[r + lh * 8] = 1.0f / lrun[r];
    }
    __syncthreads();
    float linv = mySt[l15];
    float* ob = out + (size_t)bh * CH * SEQ + tWave;
    #pragma unroll
    for (int ct = 0; ct < 4; ++ct)
        #pragma unroll
        for (int r = 0; r < 8; ++r) {
            int c = ct * 16 + r + lh * 8;
            ob[(size_t)c * SEQ + l15] = acc[ct][r] * linv;
        }
}

extern "C" void kernel_launch(void* const* d_in, const int* in_sizes, int n_in,
                              void* d_out, int out_size, void* d_ws, size_t ws_size,
                              hipStream_t stream) {
    const float* qkv = (const float*)d_in[0];
    float* out = (float*)d_out;
    dim3 grid(SEQ / BT, 64);   // 16 t-blocks x 64 (bs*heads)
    dim3 block(256);
    qkv_attn_flash_f16wmma<<<grid, block, 0, stream>>>(qkv, out);
}
